// Encoder_44135083933971
// MI455X (gfx1250) — compile-verified, matrix operand order checked by reading, not run on hardware
//
#include <hip/hip_runtime.h>
#include <math.h>

typedef __attribute__((ext_vector_type(16))) __bf16 v16bf;
typedef __attribute__((ext_vector_type(8)))  float  v8f;

#define N_NODES 50000

// ---------------------------------------------------------------------------
// Pack W [K x N] (f32, row-major) into WMMA B-operand stripes (bf16):
// layout [ntile][kstep][lane][16], each lane's 16 bf16 = contiguous 32 bytes.
// Lane l (l<16): col n = nt*16+l, K = ks*32 + {0..7, 16..23}
// Lane l (>=16): col n = nt*16+(l-16), K = ks*32 + {8..15, 24..31}
// ---------------------------------------------------------------------------
__global__ void pack_w_bf16(const float* __restrict__ W, __bf16* __restrict__ out,
                            int K, int N) {
  int idx = blockIdx.x * blockDim.x + threadIdx.x;
  if (idx >= K * N) return;
  int j    = idx & 15;
  int lane = (idx >> 4) & 31;
  int blk  = idx >> 9;          // (nt * ksteps + ks)
  int ksteps = K >> 5;
  int ks = blk % ksteps;
  int nt = blk / ksteps;
  int n    = nt * 16 + (lane & 15);
  int koff = (lane >> 4) * 8;
  int k    = ks * 32 + koff + (j < 8 ? j : j + 8);
  out[idx] = (__bf16)W[(size_t)k * N + n];
}

// ---------------------------------------------------------------------------
// BF16 WMMA GEMM: C[M x N] = A[M x K] * B, A is f32 row-major (converted on
// the fly), B is pre-packed bf16. One 16-row block per workgroup, one 16-col
// tile per wave. Requires M % 16 == 0, K % 32 == 0, blockDim.x == (N/16)*32.
// ---------------------------------------------------------------------------
__global__ void gemm_wmma_bf16(const float* __restrict__ A,
                               const __bf16* __restrict__ Bp,
                               float* __restrict__ C,
                               int K, int N) {
  const int lane = threadIdx.x & 31;
  const int wave = threadIdx.x >> 5;            // n-tile index
  const int r0   = blockIdx.x * 16;
  const int m    = lane & 15;
  const int koff = (lane >> 4) * 8;
  const int ksteps = K >> 5;

  const float*  arow = A  + (size_t)(r0 + m) * K + koff;
  const __bf16* bpt  = Bp + ((size_t)wave * ksteps * 32 + lane) * 16;

  v8f acc = {};
  for (int ks = 0; ks < ksteps; ++ks) {
    const float* ap = arow + ks * 32;
    v16bf a;
#pragma unroll
    for (int j = 0; j < 8; ++j) {
      a[j]     = (__bf16)ap[j];        // K = ks*32 + koff + j
      a[j + 8] = (__bf16)ap[16 + j];   // K = ks*32 + koff + 16 + j
    }
    v16bf b = *(const v16bf*)(bpt + (size_t)ks * 32 * 16);
    acc = __builtin_amdgcn_wmma_f32_16x16x32_bf16(
        /*neg_a=*/false, a, /*neg_b=*/false, b,
        /*c_mod=*/(short)0, acc, /*reuse_a=*/false, /*reuse_b=*/false);
  }

  // C/D layout: VGPR r -> M = r (lanes 0-15) or 8+r (lanes 16-31); N = lane&15
  const int n     = wave * 16 + (lane & 15);
  const int mbase = r0 + ((lane >> 4) << 3);
#pragma unroll
  for (int r = 0; r < 8; ++r)
    C[(size_t)(mbase + r) * N + n] = acc[r];
}

// ---------------------------------------------------------------------------
// Degree / normalization
// ---------------------------------------------------------------------------
__global__ void deg_init(float* deg, int n) {
  int i = blockIdx.x * blockDim.x + threadIdx.x;
  if (i < n) deg[i] = 1.0f;                       // self-loop contribution
}
__global__ void deg_edges(const int* __restrict__ dst, int E, float* deg) {
  int e = blockIdx.x * blockDim.x + threadIdx.x;
  if (e < E) unsafeAtomicAdd(&deg[dst[e]], 1.0f); // global_atomic_add_f32
}
__global__ void deg_to_dinv(float* deg, int n) {
  int i = blockIdx.x * blockDim.x + threadIdx.x;
  if (i < n) { float d = deg[i]; deg[i] = d > 0.0f ? rsqrtf(d) : 0.0f; }
}

// out[n,f] = dinv[n]^2 * h[n,f]   (self-loop term initializes the accumulator)
__global__ void selfloop_init(const float* __restrict__ h,
                              const float* __restrict__ dinv,
                              float* __restrict__ out, int F, size_t total) {
  size_t i = (size_t)blockIdx.x * blockDim.x + threadIdx.x;
  if (i >= total) return;
  int node = (int)(i / F);
  float dv = dinv[node];
  out[i] = dv * dv * h[i];
}

// one wave per edge: out[dst] += dinv[src]*dinv[dst] * h[src]
__global__ void edge_scatter(const int* __restrict__ src,
                             const int* __restrict__ dst, int E,
                             const float* __restrict__ dinv,
                             const float* __restrict__ h,
                             float* __restrict__ out, int F) {
  int e    = blockIdx.x * (blockDim.x >> 5) + (threadIdx.x >> 5);
  int lane = threadIdx.x & 31;
  if (e >= E) return;
  int s = src[e], d = dst[e];
  float w = dinv[s] * dinv[d];
  const float* hs = h + (size_t)s * F;
  float* od = out + (size_t)d * F;
  for (int f = lane; f < F; f += 32)
    unsafeAtomicAdd(&od[f], w * hs[f]);
}

__global__ void bias_relu(const float* __restrict__ in, const float* __restrict__ b,
                          float* __restrict__ out, size_t total, int F) {
  size_t i = (size_t)blockIdx.x * blockDim.x + threadIdx.x;
  if (i >= total) return;
  float v = in[i] + b[i % F];
  out[i] = v > 0.0f ? v : 0.0f;
}

// ---------------------------------------------------------------------------
extern "C" void kernel_launch(void* const* d_in, const int* in_sizes, int n_in,
                              void* d_out, int out_size, void* d_ws, size_t ws_size,
                              hipStream_t stream) {
  const float* x  = (const float*)d_in[0];   // [50000, 512]
  const int*   ei = (const int*)d_in[1];     // [2, E]
  const float* W1 = (const float*)d_in[2];   // [512, 128]
  const float* b1 = (const float*)d_in[3];   // [128]
  const float* W2 = (const float*)d_in[4];   // [128, 64]
  const float* b2 = (const float*)d_in[5];   // [64]
  float* out = (float*)d_out;                // [50000, 64]

  const int E = in_sizes[1] / 2;
  const int* src = ei;
  const int* dst = ei + E;

  // --- workspace carve-up (256B aligned) ---
  char* p = (char*)d_ws;
  auto take = [&p](size_t bytes) {
    char* r = p; p += (bytes + 255) & ~(size_t)255; return r;
  };
  float*  dinv = (float*)take((size_t)N_NODES * 4);
  float*  h1   = (float*)take((size_t)N_NODES * 128 * 4);
  float*  agg1 = (float*)take((size_t)N_NODES * 128 * 4);
  float*  h2   = (float*)take((size_t)N_NODES * 64 * 4);
  float*  agg2 = (float*)take((size_t)N_NODES * 64 * 4);
  __bf16* pw1  = (__bf16*)take((size_t)512 * 128 * 2);
  __bf16* pw2  = (__bf16*)take((size_t)128 * 64 * 2);

  const int TB = 256;
  const size_t n1 = (size_t)N_NODES * 128;   // 6.4M
  const size_t n2 = (size_t)N_NODES * 64;    // 3.2M

  // Pack weights into WMMA B-operand layout (bf16)
  pack_w_bf16<<<(512 * 128 + TB - 1) / TB, TB, 0, stream>>>(W1, pw1, 512, 128);
  pack_w_bf16<<<(128 * 64  + TB - 1) / TB, TB, 0, stream>>>(W2, pw2, 128, 64);

  // Symmetric normalization coefficients
  deg_init   <<<(N_NODES + TB - 1) / TB, TB, 0, stream>>>(dinv, N_NODES);
  deg_edges  <<<(E + TB - 1) / TB, TB, 0, stream>>>(dst, E, dinv);
  deg_to_dinv<<<(N_NODES + TB - 1) / TB, TB, 0, stream>>>(dinv, N_NODES);

  // Layer 1: h1 = x @ W1  (WMMA bf16, f32 accumulate)
  gemm_wmma_bf16<<<N_NODES / 16, 8 * 32, 0, stream>>>(x, pw1, h1, 512, 128);
  selfloop_init<<<(n1 + TB - 1) / TB, TB, 0, stream>>>(h1, dinv, agg1, 128, n1);
  edge_scatter <<<(E + 7) / 8, TB, 0, stream>>>(src, dst, E, dinv, h1, agg1, 128);
  bias_relu    <<<(n1 + TB - 1) / TB, TB, 0, stream>>>(agg1, b1, agg1, n1, 128);

  // Layer 2: h2 = relu(agg1) @ W2
  gemm_wmma_bf16<<<N_NODES / 16, 4 * 32, 0, stream>>>(agg1, pw2, h2, 128, 64);
  selfloop_init<<<(n2 + TB - 1) / TB, TB, 0, stream>>>(h2, dinv, agg2, 64, n2);
  edge_scatter <<<(E + 7) / 8, TB, 0, stream>>>(src, dst, E, dinv, h2, agg2, 64);
  bias_relu    <<<(n2 + TB - 1) / TB, TB, 0, stream>>>(agg2, b2, out, n2, 64);
}
